// MultiHeadAttention_773094114067
// MI455X (gfx1250) — matrix operand order repaired and checked by reading, not compiled, not run on hardware
//
#include <hip/hip_runtime.h>

// ---------------------------------------------------------------------------
// MHA for MI455X (gfx1250): bf16 WMMA GEMMs + flash attention, wave32.
//   B=2, S=2048, D_MODEL=768, H=12, DK=64
// GEMM: block tile 256x128, 8 waves, wave tile 64x64 (16 WMMA / 16 ds_load),
//       k-step 32, register double-buffered global->LDS staging.
// Attention: 4 waves x 16 q-rows, kv-block 64, online softmax, P via LDS,
//       register double-buffered K/V staging + L2 prefetch of block jb+2.
// ---------------------------------------------------------------------------

#define BATCH   2
#define SEQ     2048
#define DM      768
#define NH      12
#define DK      64
#define M_TOT   (BATCH * SEQ)     // 4096

typedef __attribute__((ext_vector_type(16))) __bf16 v16bf;
typedef __attribute__((ext_vector_type(8)))  float  v8f;

union FragAB { v16bf v; uint4 q[2]; };      // 32B: one WMMA A or B operand

static __device__ __forceinline__ unsigned short f32_to_bf16(float f) {
    unsigned int u = __float_as_uint(f);
    u += 0x7FFFu + ((u >> 16) & 1u);        // round-to-nearest-even
    return (unsigned short)(u >> 16);
}

static __device__ __forceinline__ float half16_max(float v) {
    v = fmaxf(v, __shfl_xor(v, 1, 32));
    v = fmaxf(v, __shfl_xor(v, 2, 32));
    v = fmaxf(v, __shfl_xor(v, 4, 32));
    v = fmaxf(v, __shfl_xor(v, 8, 32));
    return v;
}
static __device__ __forceinline__ float half16_sum(float v) {
    v += __shfl_xor(v, 1, 32);
    v += __shfl_xor(v, 2, 32);
    v += __shfl_xor(v, 4, 32);
    v += __shfl_xor(v, 8, 32);
    return v;
}

// ---------------------------------------------------------------------------
// Weight convert: W[k][n] fp32 -> Wt[n][k] bf16  (B operand wants [N][K])
// ---------------------------------------------------------------------------
__global__ void wt_convert(const float* __restrict__ W, unsigned short* __restrict__ Wt) {
    int i = blockIdx.x * 256 + threadIdx.x;   // 0 .. DM*DM-1
    int n = i / DM, k = i % DM;
    Wt[i] = f32_to_bf16(W[k * DM + n]);
}

// ---------------------------------------------------------------------------
// Tiled WMMA GEMM:  C[M_TOT x DM] = A[M_TOT x DM] * Wt^T + bias
// Epilogues:
//   EPI_QK : bf16 out, split-heads layout [B,H,S,DK]
//   EPI_VT : bf16 out, transposed heads   [B,H,DK,S]  (for PV B-operand)
//   EPI_F32: fp32 out, row-major [M_TOT, DM]          (final projection)
// ---------------------------------------------------------------------------
enum { EPI_QK = 0, EPI_VT = 1, EPI_F32 = 2 };

template <int EPI, bool ABF16>
__global__ __launch_bounds__(256)
void gemm_wmma(const void* __restrict__ Ap,
               const unsigned short* __restrict__ Bt,   // [N][K] bf16
               const float* __restrict__ bias,
               void* __restrict__ Cp)
{
    __shared__ __align__(16) unsigned short As[256 * 32];   // [m][k]  16 KB
    __shared__ __align__(16) unsigned short Bs[128 * 32];   // [n][k]   8 KB

    const int m0 = blockIdx.y * 256;
    const int n0 = blockIdx.x * 128;
    const int t    = threadIdx.x;
    const int lane = t & 31;
    const int w    = t >> 5;
    const int wm   = w & 3;        // 4 wave rows  (64 rows each)
    const int wn   = w >> 2;       // 2 wave cols  (64 cols each)
    const int h    = lane >> 4;
    const int r    = lane & 15;

    v8f acc[4][4];
#pragma unroll
    for (int im = 0; im < 4; ++im)
#pragma unroll
        for (int in = 0; in < 4; ++in)
            acc[im][in] = (v8f){0.f,0.f,0.f,0.f,0.f,0.f,0.f,0.f};

    // staging: A = one 32-elem row per thread; B = 2 threads per 32-elem row
    const int arow  = t;
    const int brow  = t >> 1, bhalf = t & 1;

    float4 pa_f[8];      // fp32 A prefetch (32 floats)
    uint4  pa_b[4];      // bf16 A prefetch (32 bf16)
    uint4  pb[2];        // B prefetch (16 bf16)

    // ---- prefetch k-step 0 ----
    {
        if (ABF16) {
            const uint4* src = (const uint4*)((const unsigned short*)Ap + (size_t)(m0 + arow) * DM);
#pragma unroll
            for (int i = 0; i < 4; ++i) pa_b[i] = src[i];
        } else {
            const float4* src = (const float4*)((const float*)Ap + (size_t)(m0 + arow) * DM);
#pragma unroll
            for (int i = 0; i < 8; ++i) pa_f[i] = src[i];
        }
        const uint4* sb = (const uint4*)(Bt + (size_t)(n0 + brow) * DM + bhalf * 16);
        pb[0] = sb[0]; pb[1] = sb[1];
    }

    for (int kk = 0; kk < DM / 32; ++kk) {
        if (kk) __syncthreads();            // all waves done reading prev tiles
        // ---- store prefetched tile to LDS ----
        if (ABF16) {
            uint4* dst = (uint4*)&As[arow * 32];
#pragma unroll
            for (int i = 0; i < 4; ++i) dst[i] = pa_b[i];
        } else {
            union { unsigned short s16[32]; uint4 qv[4]; } tmp;
#pragma unroll
            for (int i = 0; i < 8; ++i) {
                float4 f = pa_f[i];
                tmp.s16[i*4+0] = f32_to_bf16(f.x);
                tmp.s16[i*4+1] = f32_to_bf16(f.y);
                tmp.s16[i*4+2] = f32_to_bf16(f.z);
                tmp.s16[i*4+3] = f32_to_bf16(f.w);
            }
            uint4* dst = (uint4*)&As[arow * 32];
#pragma unroll
            for (int i = 0; i < 4; ++i) dst[i] = tmp.qv[i];
        }
        {
            uint4* dst = (uint4*)&Bs[brow * 32 + bhalf * 16];
            dst[0] = pb[0]; dst[1] = pb[1];
        }
        __syncthreads();

        // ---- prefetch next k-step while computing this one ----
        if (kk + 1 < DM / 32) {
            const int kb = (kk + 1) * 32;
            if (ABF16) {
                const uint4* src = (const uint4*)((const unsigned short*)Ap + (size_t)(m0 + arow) * DM + kb);
#pragma unroll
                for (int i = 0; i < 4; ++i) pa_b[i] = src[i];
            } else {
                const float4* src = (const float4*)((const float*)Ap + (size_t)(m0 + arow) * DM + kb);
#pragma unroll
                for (int i = 0; i < 8; ++i) pa_f[i] = src[i];
            }
            const uint4* sb = (const uint4*)(Bt + (size_t)(n0 + brow) * DM + kb + bhalf * 16);
            pb[0] = sb[0]; pb[1] = sb[1];
        }

        // ---- fragments per documented layouts ----
        FragAB a[4], b[4];
#pragma unroll
        for (int im = 0; im < 4; ++im) {
            int row = wm * 64 + im * 16 + r;
            a[im].q[0] = *(const uint4*)&As[row * 32 + 8 * h];        // K = 8h..8h+7
            a[im].q[1] = *(const uint4*)&As[row * 32 + 16 + 8 * h];   // K = 16+8h..
        }
#pragma unroll
        for (int in = 0; in < 4; ++in) {
            int col = wn * 64 + in * 16 + r;
            b[in].q[0] = *(const uint4*)&Bs[col * 32 + 8 * h];        // K = 8h..8h+7
            b[in].q[1] = *(const uint4*)&Bs[col * 32 + 16 + 8 * h];   // K = 16+8h..
        }
#pragma unroll
        for (int im = 0; im < 4; ++im)
#pragma unroll
            for (int in = 0; in < 4; ++in)
                acc[im][in] = __builtin_amdgcn_wmma_f32_16x16x32_bf16(
                    false, a[im].v, false, b[in].v, (short)0, acc[im][in], false, false);
    }

    // ---- epilogue ----
#pragma unroll
    for (int in = 0; in < 4; ++in) {
        const int n = n0 + wn * 64 + in * 16 + r;
        const float bv = bias[n];
#pragma unroll
        for (int im = 0; im < 4; ++im) {
#pragma unroll
            for (int j = 0; j < 8; ++j) {
                const int m = m0 + wm * 64 + im * 16 + 8 * h + j;
                const float val = acc[im][in][j] + bv;
                if (EPI == EPI_F32) {
                    ((float*)Cp)[(size_t)m * DM + n] = val;
                } else {
                    const int bb = m >> 11, s = m & (SEQ - 1);
                    const int hd = n >> 6,  d = n & (DK - 1);
                    unsigned short o = f32_to_bf16(val);
                    if (EPI == EPI_QK)
                        ((unsigned short*)Cp)[(((size_t)(bb * NH + hd) * SEQ + s) << 6) + d] = o;
                    else // EPI_VT
                        ((unsigned short*)Cp)[((size_t)(bb * NH + hd) * DK + d) * SEQ + s] = o;
                }
            }
        }
    }
}

// ---------------------------------------------------------------------------
// Flash attention: one WG = 64 query rows for one (b,h); 128 threads = 4 waves.
// Q[B,H,S,DK] bf16, K[B,H,S,DK] bf16, Vt[B,H,DK,S] bf16 -> AO[B,S,H*DK] bf16.
// ---------------------------------------------------------------------------
__global__ __launch_bounds__(128)
void attn_flash(const unsigned short* __restrict__ Qg,
                const unsigned short* __restrict__ Kg,
                const unsigned short* __restrict__ Vtg,
                unsigned short* __restrict__ AO)
{
    __shared__ __align__(16) unsigned short Qs [64 * 64];
    __shared__ __align__(16) unsigned short Ks [64 * 64];
    __shared__ __align__(16) unsigned short Vts[64 * 64];   // [dv][kv]
    __shared__ __align__(16) unsigned short Ps [64 * 64];   // [q][kv] bf16

    const int qb = blockIdx.x;            // 0..31 query block
    const int bh = blockIdx.y;            // 0..23 (b*NH + h)
    const int t    = threadIdx.x;
    const int lane = t & 31;
    const int w    = t >> 5;
    const int h    = lane >> 4;
    const int r    = lane & 15;
    const int q0   = qb * 64;
    const size_t headQK = (size_t)bh * SEQ * DK;
    const size_t headVT = (size_t)bh * DK * SEQ;

    const int rr = t >> 1, hf = t & 1;    // staging: 2 threads per 128B row

    // ---- stage Q block once (64 rows x 128B; 64B per thread) ----
    {
        const uint4* src = (const uint4*)(Qg + headQK + (size_t)(q0 + rr) * DK + hf * 32);
        uint4* dst = (uint4*)&Qs[rr * 64 + hf * 32];
#pragma unroll
        for (int i = 0; i < 4; ++i) dst[i] = src[i];
    }
    __syncthreads();

    FragAB qa[2];
#pragma unroll
    for (int ks = 0; ks < 2; ++ks) {
        const int row = w * 16 + r;
        qa[ks].q[0] = *(const uint4*)&Qs[row * 64 + ks * 32 + 8 * h];
        qa[ks].q[1] = *(const uint4*)&Qs[row * 64 + ks * 32 + 16 + 8 * h];
    }

    float m_run[8], l_run[8];
    v8f o[4];
#pragma unroll
    for (int j = 0; j < 8; ++j) { m_run[j] = -1e30f; l_run[j] = 0.f; }
#pragma unroll
    for (int tn = 0; tn < 4; ++tn) o[tn] = (v8f){0.f,0.f,0.f,0.f,0.f,0.f,0.f,0.f};

    // ---- K/V register prefetch of block 0 ----
    uint4 pk[4], pv[4];
    {
        const uint4* sk = (const uint4*)(Kg + headQK + (size_t)rr * DK + hf * 32);
        const uint4* sv = (const uint4*)(Vtg + headVT + (size_t)rr * SEQ + hf * 32);
#pragma unroll
        for (int i = 0; i < 4; ++i) { pk[i] = sk[i]; pv[i] = sv[i]; }
    }

    for (int jb = 0; jb < SEQ / 64; ++jb) {
        if (jb) __syncthreads();          // all waves done reading prev K/V
        {
            uint4* dk = (uint4*)&Ks[rr * 64 + hf * 32];
            uint4* dv = (uint4*)&Vts[rr * 64 + hf * 32];
#pragma unroll
            for (int i = 0; i < 4; ++i) { dk[i] = pk[i]; dv[i] = pv[i]; }
        }
        __syncthreads();

        // prefetch next block into regs; warm L2 for jb+2
        if (jb + 1 < SEQ / 64) {
            const uint4* sk = (const uint4*)(Kg + headQK + (size_t)((jb + 1) * 64 + rr) * DK + hf * 32);
            const uint4* sv = (const uint4*)(Vtg + headVT + (size_t)rr * SEQ + (jb + 1) * 64 + hf * 32);
#pragma unroll
            for (int i = 0; i < 4; ++i) { pk[i] = sk[i]; pv[i] = sv[i]; }
        }
        if (jb + 2 < SEQ / 64) {
            __builtin_prefetch(Kg + headQK + (size_t)((jb + 2) * 64 + rr) * DK, 0, 1);
            __builtin_prefetch(Vtg + headVT + (size_t)rr * SEQ + (jb + 2) * 64, 0, 1);
        }

        // ---- S = Q K^T (16 q rows x 64 kv cols per wave) ----
        v8f s[4];
#pragma unroll
        for (int tn = 0; tn < 4; ++tn) s[tn] = (v8f){0.f,0.f,0.f,0.f,0.f,0.f,0.f,0.f};
#pragma unroll
        for (int ks = 0; ks < 2; ++ks) {
#pragma unroll
            for (int tn = 0; tn < 4; ++tn) {
                FragAB kb;
                const int col = tn * 16 + r;
                kb.q[0] = *(const uint4*)&Ks[col * 64 + ks * 32 + 8 * h];
                kb.q[1] = *(const uint4*)&Ks[col * 64 + ks * 32 + 16 + 8 * h];
                s[tn] = __builtin_amdgcn_wmma_f32_16x16x32_bf16(
                    false, qa[ks].v, false, kb.v, (short)0, s[tn], false, false);
            }
        }

        // ---- online softmax (row = 8h + j within wave's 16 rows) ----
#pragma unroll
        for (int j = 0; j < 8; ++j) {
            float rm = -1e30f;
#pragma unroll
            for (int tn = 0; tn < 4; ++tn) {
                s[tn][j] *= 0.125f;                     // 1/sqrt(DK)
                rm = fmaxf(rm, s[tn][j]);
            }
            rm = half16_max(rm);
            const float mn = fmaxf(m_run[j], rm);
            const float sc = __expf(m_run[j] - mn);
            float rs = 0.f;
#pragma unroll
            for (int tn = 0; tn < 4; ++tn) {
                float p = __expf(s[tn][j] - mn);
                s[tn][j] = p;
                rs += p;
            }
            rs = half16_sum(rs);
            l_run[j] = l_run[j] * sc + rs;
            m_run[j] = mn;
#pragma unroll
            for (int tn = 0; tn < 4; ++tn) o[tn][j] *= sc;
            // P -> LDS (bf16), own wave's rows only: no cross-wave barrier needed
            const int row = w * 16 + 8 * h + j;
#pragma unroll
            for (int tn = 0; tn < 4; ++tn)
                Ps[row * 64 + tn * 16 + r] = f32_to_bf16(s[tn][j]);
        }

        // ---- O += P V  (contraction over kv=64) ----
#pragma unroll
        for (int ks = 0; ks < 2; ++ks) {
            FragAB pa;
            const int row = w * 16 + r;
            pa.q[0] = *(const uint4*)&Ps[row * 64 + ks * 32 + 8 * h];
            pa.q[1] = *(const uint4*)&Ps[row * 64 + ks * 32 + 16 + 8 * h];
#pragma unroll
            for (int tn = 0; tn < 4; ++tn) {
                FragAB vb;
                const int col = tn * 16 + r;                 // dv column
                vb.q[0] = *(const uint4*)&Vts[col * 64 + ks * 32 + 8 * h];
                vb.q[1] = *(const uint4*)&Vts[col * 64 + ks * 32 + 16 + 8 * h];
                o[tn] = __builtin_amdgcn_wmma_f32_16x16x32_bf16(
                    false, pa.v, false, vb.v, (short)0, o[tn], false, false);
            }
        }
    }

    // ---- epilogue: normalize, write concat-heads bf16 [B,S,DM] ----
    const int bb = bh / NH, hd = bh % NH;
#pragma unroll
    for (int j = 0; j < 8; ++j) {
        const float inv = 1.f / l_run[j];
        const int srow = q0 + w * 16 + 8 * h + j;
#pragma unroll
        for (int tn = 0; tn < 4; ++tn)
            AO[((size_t)(bb * SEQ + srow)) * DM + hd * DK + tn * 16 + r] =
                f32_to_bf16(o[tn][j] * inv);
    }
}

// ---------------------------------------------------------------------------
extern "C" void kernel_launch(void* const* d_in, const int* in_sizes, int n_in,
                              void* d_out, int out_size, void* d_ws, size_t ws_size,
                              hipStream_t stream) {
    const float* q_in = (const float*)d_in[0];
    const float* k_in = (const float*)d_in[1];
    const float* v_in = (const float*)d_in[2];
    const float* Wq = (const float*)d_in[3];  const float* bq = (const float*)d_in[4];
    const float* Wk = (const float*)d_in[5];  const float* bk = (const float*)d_in[6];
    const float* Wv = (const float*)d_in[7];  const float* bv = (const float*)d_in[8];
    const float* Wo = (const float*)d_in[9];  const float* bo = (const float*)d_in[10];

    char* ws = (char*)d_ws;
    const size_t WT = (size_t)DM * DM * 2;                 // bf16 weight: 1.125 MB
    const size_t QS = (size_t)BATCH * NH * SEQ * DK * 2;   // 6 MB
    unsigned short* wtq = (unsigned short*)(ws);
    unsigned short* wtk = (unsigned short*)(ws + WT);
    unsigned short* wtv = (unsigned short*)(ws + 2 * WT);
    unsigned short* wto = (unsigned short*)(ws + 3 * WT);
    unsigned short* Qg  = (unsigned short*)(ws + 4 * WT);
    unsigned short* Kg  = (unsigned short*)(ws + 4 * WT + QS);
    unsigned short* Vtg = (unsigned short*)(ws + 4 * WT + 2 * QS);
    unsigned short* AOg = (unsigned short*)(ws + 4 * WT + 3 * QS);

    const int cvt_blocks = (DM * DM) / 256;
    wt_convert<<<cvt_blocks, 256, 0, stream>>>(Wq, wtq);
    wt_convert<<<cvt_blocks, 256, 0, stream>>>(Wk, wtk);
    wt_convert<<<cvt_blocks, 256, 0, stream>>>(Wv, wtv);
    wt_convert<<<cvt_blocks, 256, 0, stream>>>(Wo, wto);

    dim3 gg(DM / 128, M_TOT / 256);   // (6, 16)
    gemm_wmma<EPI_QK,  false><<<gg, 256, 0, stream>>>(q_in, wtq, bq, Qg);
    gemm_wmma<EPI_QK,  false><<<gg, 256, 0, stream>>>(k_in, wtk, bk, Kg);
    gemm_wmma<EPI_VT,  false><<<gg, 256, 0, stream>>>(v_in, wtv, bv, Vtg);

    attn_flash<<<dim3(SEQ / 64, BATCH * NH), 128, 0, stream>>>(Qg, Kg, Vtg, AOg);

    gemm_wmma<EPI_F32, true ><<<gg, 256, 0, stream>>>(AOg, wto, bo, (float*)d_out);
}
